// MultiHeadSelfAttention_57200374448239
// MI455X (gfx1250) — matrix operand order repaired and checked by reading, not compiled, not run on hardware
//
#include <hip/hip_runtime.h>
#include <hip/hip_bf16.h>

// ---------------------------------------------------------------------------
// Problem constants (match reference)
// ---------------------------------------------------------------------------
#define BB 2
#define SS 2048
#define DM 1024
#define HH 16
#define DK 64
#define E3 3072                 // 3*DM
#define ROWS (BB*SS)            // 4096
#define BH (BB*HH)              // 32

typedef __attribute__((ext_vector_type(16))) _Float16 v16h;
typedef __attribute__((ext_vector_type(8)))  _Float16 v8h;
typedef __attribute__((ext_vector_type(8)))  float    v8f;
typedef int v4i __attribute__((vector_size(16)));

// ---------------------------------------------------------------------------
// Async global->LDS staging (gfx1250 GLOBAL_LOAD_ASYNC_TO_LDS_B128, ASYNCcnt)
// with a portable synchronous fallback.
// ---------------------------------------------------------------------------
#if __has_builtin(__builtin_amdgcn_global_load_async_to_lds_b128)
#define HAVE_ASYNC_LDS 1
__device__ __forceinline__ void cp16_async(_Float16* lds, const _Float16* g) {
    __builtin_amdgcn_global_load_async_to_lds_b128(
        (__attribute__((address_space(1))) v4i*)g,
        (__attribute__((address_space(3))) v4i*)lds, 0, 0);
}
__device__ __forceinline__ void wait_async_lds() {
#if __has_builtin(__builtin_amdgcn_s_wait_asynccnt)
    __builtin_amdgcn_s_wait_asynccnt(0);
#else
    asm volatile("s_wait_asynccnt 0" ::: "memory");
#endif
}
#else
#define HAVE_ASYNC_LDS 0
__device__ __forceinline__ void cp16_async(_Float16* lds, const _Float16* g) {
    *(v8h*)lds = *(const v8h*)g;
}
__device__ __forceinline__ void wait_async_lds() {}
#endif

// ---------------------------------------------------------------------------
// WMMA helper: D = A(16x32 f16) * B(32x16 f16) + C(16x16 f32)
// ---------------------------------------------------------------------------
__device__ __forceinline__ v8f wmma_f16(v16h a, v16h b, v8f c) {
    return __builtin_amdgcn_wmma_f32_16x16x32_f16(
        /*neg_a=*/false, a, /*neg_b=*/false, b,
        /*c_mod=*/(short)0, c, /*reuse_a=*/false, /*reuse_b=*/false);
}

// A fragment (16x32, f16) from row-major matrix at (row0, k0).
// lanes 0-15: row M=lane, K=0..15; lanes 16-31: same M, K=16..31
__device__ __forceinline__ v16h frag_a_load(const _Float16* __restrict__ base,
                                            int ld, int row0, int k0) {
    int l   = threadIdx.x & 31;
    int m   = l & 15;
    int off = (l >> 4) << 4;            // 0 or 16
    const _Float16* p = base + (size_t)(row0 + m) * ld + k0 + off;
    v8h lo = *(const v8h*)(p);
    v8h hi = *(const v8h*)(p + 8);
    return __builtin_shufflevector(lo, hi, 0,1,2,3,4,5,6,7,8,9,10,11,12,13,14,15);
}

// B fragment (32x16, f16): logical B[k][n] = M[n0+n][k0+k], M row-major ld.
// lanes 0-15 hold K=0..15 for N=lane; lanes 16-31 hold K=16..31 for N=lane-16.
__device__ __forceinline__ v16h frag_bt_load(const _Float16* __restrict__ base,
                                             int ld, int n0, int k0) {
    int l    = threadIdx.x & 31;
    int n    = l & 15;
    int koff = (l >> 4) << 4;           // 0 or 16
    const _Float16* p = base + (size_t)(n0 + n) * ld + k0 + koff;
    return *(const v16h*)(p);
}

// stage one 32-key K tile (32x64) and V^T tile (64x32) into LDS buffers
__device__ __forceinline__ void stage_kv(_Float16* KshBuf, _Float16* VshBuf,
                                         const _Float16* Kc, const _Float16* Vt,
                                         int kk, int t, int vd, int vc) {
    const _Float16* gk = Kc + (size_t)kk * DK + t * 16;
    _Float16*       lk = KshBuf + t * 16;
    const _Float16* gv = Vt + (size_t)vd * SS + kk + vc;
    _Float16*       lv = VshBuf + vd * 32 + vc;
    cp16_async(lk,     gk);
    cp16_async(lk + 8, gk + 8);
    cp16_async(lv,     gv);
    cp16_async(lv + 8, gv + 8);
}

// ---------------------------------------------------------------------------
// Kernel 1: f32 -> f16 conversion
// ---------------------------------------------------------------------------
__global__ __launch_bounds__(256)
void f32_to_f16_kernel(const float* __restrict__ in, _Float16* __restrict__ out, int n) {
    int i = blockIdx.x * 256 + threadIdx.x;
    int stride = gridDim.x * 256;
    for (; i < n; i += stride) out[i] = (_Float16)in[i];
}

// ---------------------------------------------------------------------------
// Kernel 2: C[M][N] = A[M][K] * W[N][K]^T   (f16 in, f32 out)
// one wave per 64x64 C tile: 4 A-frags x 4 B-frags -> 16 WMMAs per 32-k step
// ---------------------------------------------------------------------------
__global__ __launch_bounds__(128)
void gemm_nt_kernel(const _Float16* __restrict__ A, const _Float16* __restrict__ W,
                    float* __restrict__ C, int M, int N, int K) {
    int wave = blockIdx.x * (blockDim.x >> 5) + (threadIdx.x >> 5);
    int ntn  = N >> 6;                  // 64-wide N tiles
    int row0 = (wave / ntn) << 6;
    int n0   = (wave % ntn) << 6;

    v8f acc[4][4];
#pragma unroll
    for (int i = 0; i < 4; ++i)
#pragma unroll
        for (int j = 0; j < 4; ++j) acc[i][j] = (v8f){};

    for (int k0 = 0; k0 < K; k0 += 32) {
        v16h a0 = frag_a_load(A, K, row0,      k0);
        v16h a1 = frag_a_load(A, K, row0 + 16, k0);
        v16h a2 = frag_a_load(A, K, row0 + 32, k0);
        v16h a3 = frag_a_load(A, K, row0 + 48, k0);
#pragma unroll
        for (int j = 0; j < 4; ++j) {
            v16h b = frag_bt_load(W, K, n0 + 16 * j, k0);
            acc[0][j] = wmma_f16(a0, b, acc[0][j]);
            acc[1][j] = wmma_f16(a1, b, acc[1][j]);
            acc[2][j] = wmma_f16(a2, b, acc[2][j]);
            acc[3][j] = wmma_f16(a3, b, acc[3][j]);
        }
    }

    int l  = threadIdx.x & 31;
    int n  = l & 15;
    int mh = (l >> 4) << 3;
#pragma unroll
    for (int i = 0; i < 4; ++i)
#pragma unroll
        for (int j = 0; j < 4; ++j)
#pragma unroll
            for (int r = 0; r < 8; ++r)
                C[(size_t)(row0 + 16 * i + r + mh) * N + n0 + 16 * j + n] = acc[i][j][r];
}

// ---------------------------------------------------------------------------
// Kernel 3: RoPE + head split.  qkv f32 [ROWS][3072] ->
//   qh,kh f16 [BH][S][64] (rotated), vT f16 [BH][64][S]
// ---------------------------------------------------------------------------
__global__ __launch_bounds__(256)
void rope_split_kernel(const float* __restrict__ qkv, const int* __restrict__ pos,
                       _Float16* __restrict__ qh, _Float16* __restrict__ kh,
                       _Float16* __restrict__ vT) {
    int idx = blockIdx.x * 256 + threadIdx.x;
    if (idx >= BB * SS * HH * (DK / 2)) return;
    int i  = idx & 31;                  // pair index 0..31
    int t  = idx >> 5;
    int h  = t % HH;
    int bs = t / HH;
    int s  = bs % SS;
    int b  = bs / SS;
    int row = b * SS + s;

    float p = (float)pos[row];
    const float LN_THETA = 9.210340371976184f;
    float inv = __expf(-(float)(2 * i) * (LN_THETA / 64.0f));   // THETA^(-2i/64)
    float ang = p * inv;
    float sn, cs;
    __sincosf(ang, &sn, &cs);

    const float* qrow = qkv + (size_t)row * E3;
    size_t o = ((size_t)(b * HH + h) * SS + s) * DK + 2 * i;

    float q1 = qrow[h * DK + 2 * i], q2 = qrow[h * DK + 2 * i + 1];
    qh[o]     = (_Float16)(q1 * cs - q2 * sn);
    qh[o + 1] = (_Float16)(q1 * sn + q2 * cs);

    float k1 = qrow[DM + h * DK + 2 * i], k2 = qrow[DM + h * DK + 2 * i + 1];
    kh[o]     = (_Float16)(k1 * cs - k2 * sn);
    kh[o + 1] = (_Float16)(k1 * sn + k2 * cs);

    float v1 = qrow[2 * DM + h * DK + 2 * i], v2 = qrow[2 * DM + h * DK + 2 * i + 1];
    size_t ot = ((size_t)(b * HH + h) * DK + 2 * i) * SS + s;
    vT[ot]      = (_Float16)v1;
    vT[ot + SS] = (_Float16)v2;
}

// ---------------------------------------------------------------------------
// Kernel 4: block-cooperative causal flash attention with DOUBLE-BUFFERED
// async K/V staging: while the 8 WMMAs of chunk i run, the async units pull
// chunk i+1 into the other LDS buffer (tracked on ASYNCcnt).
// One block (4 waves) owns 64 q-rows of one head.
// ---------------------------------------------------------------------------
__global__ __launch_bounds__(128)
void attn_kernel(const _Float16* __restrict__ qh, const _Float16* __restrict__ kh,
                 const _Float16* __restrict__ vT, _Float16* __restrict__ ao) {
    __shared__ __align__(16) _Float16 Ksh[2][32][64];   // [buf][key][d]  8 KB
    __shared__ __align__(16) _Float16 Vsh[2][64][32];   // [buf][d][key]  8 KB
    __shared__ __align__(16) _Float16 plds[4][16][32];  // per-wave P     4 KB

    int wid = threadIdx.x >> 5;
    int l   = threadIdx.x & 31;
    int bh  = blockIdx.x >> 5;          // S/64 = 32 blocks per head
    int Q0  = (blockIdx.x & 31) << 6;
    int q0  = Q0 + (wid << 4);          // this wave's 16 q-rows

    const _Float16* Q  = qh + (size_t)bh * SS * DK;
    const _Float16* Kc = kh + (size_t)bh * SS * DK;
    const _Float16* Vt = vT + (size_t)bh * DK * SS;

    v16h qa0 = frag_a_load(Q, DK, q0, 0);
    v16h qa1 = frag_a_load(Q, DK, q0, 32);

    v8f o0 = {}, o1 = {}, o2 = {}, o3 = {};
    float rm[8], rl[8];
#pragma unroll
    for (int r = 0; r < 8; ++r) { rm[r] = -1e30f; rl[r] = 0.0f; }

    int mh   = (l >> 4) << 3;           // C-layout row offset (0/8)
    int nn   = l & 15;                  // C-layout / B-frag column
    int koff = (l >> 4) << 4;           // B-frag K offset (0/16)

    // staging assignment: 128 threads x 32 B for each of K and V tiles
    int t  = threadIdx.x;
    int vd = t >> 1, vc = (t & 1) * 16;

    int kend = Q0 + 64;                 // uniform across the block (>= 64)

    // prologue: stage first chunk into buffer 0
    stage_kv(&Ksh[0][0][0], &Vsh[0][0][0], Kc, Vt, 0, t, vd, vc);

    for (int kk = 0; kk < kend; kk += 32) {
        int buf = (kk >> 5) & 1;

        // wait for the chunk staged last iteration, make it block-visible
        wait_async_lds();
        __syncthreads();

        // kick off async stage of the NEXT chunk into the other buffer
        if (kk + 32 < kend)
            stage_kv(&Ksh[buf ^ 1][0][0], &Vsh[buf ^ 1][0][0], Kc, Vt,
                     kk + 32, t, vd, vc);

        // ---- scores for 32 keys: two 16x16 f32 tiles ----
        v8f s0 = {}, s1 = {};
        s0 = wmma_f16(qa0, *(const v16h*)&Ksh[buf][nn][koff],           s0);
        s0 = wmma_f16(qa1, *(const v16h*)&Ksh[buf][nn][32 + koff],      s0);
        s1 = wmma_f16(qa0, *(const v16h*)&Ksh[buf][16 + nn][koff],      s1);
        s1 = wmma_f16(qa1, *(const v16h*)&Ksh[buf][16 + nn][32 + koff], s1);

        // ---- online softmax (row reductions inside 16-lane N groups) ----
#pragma unroll
        for (int r = 0; r < 8; ++r) {
            int qrow = q0 + r + mh;
            float e0 = (kk + nn      <= qrow) ? s0[r] * 0.125f : -1e30f;
            float e1 = (kk + 16 + nn <= qrow) ? s1[r] * 0.125f : -1e30f;
            float cm = fmaxf(e0, e1);
            cm = fmaxf(cm, __shfl_xor(cm, 1, 32));
            cm = fmaxf(cm, __shfl_xor(cm, 2, 32));
            cm = fmaxf(cm, __shfl_xor(cm, 4, 32));
            cm = fmaxf(cm, __shfl_xor(cm, 8, 32));
            float newm  = fmaxf(rm[r], cm);
            float alpha = __expf(rm[r] - newm);
            rm[r] = newm;
            float p0 = __expf(e0 - newm);
            float p1 = __expf(e1 - newm);
            float rs = p0 + p1;
            rs += __shfl_xor(rs, 1, 32);
            rs += __shfl_xor(rs, 2, 32);
            rs += __shfl_xor(rs, 4, 32);
            rs += __shfl_xor(rs, 8, 32);
            rl[r] = rl[r] * alpha + rs;
            o0[r] *= alpha; o1[r] *= alpha; o2[r] *= alpha; o3[r] *= alpha;
            plds[wid][r + mh][nn]      = (_Float16)p0;
            plds[wid][r + mh][16 + nn] = (_Float16)p1;
        }

        // ---- reload P as A fragment (16x32) from per-wave LDS tile ----
        {
            int m = l & 15;
            v8h lo = *(const v8h*)&plds[wid][m][2 * mh];
            v8h hi = *(const v8h*)&plds[wid][m][2 * mh + 8];
            v16h pa = __builtin_shufflevector(lo, hi,
                        0,1,2,3,4,5,6,7,8,9,10,11,12,13,14,15);
            o0 = wmma_f16(pa, *(const v16h*)&Vsh[buf][nn][koff],      o0);
            o1 = wmma_f16(pa, *(const v16h*)&Vsh[buf][16 + nn][koff], o1);
            o2 = wmma_f16(pa, *(const v16h*)&Vsh[buf][32 + nn][koff], o2);
            o3 = wmma_f16(pa, *(const v16h*)&Vsh[buf][48 + nn][koff], o3);
        }
        // no trailing barrier needed: the buffer written in iteration i+1 was
        // last read in iteration i, and the top-of-loop barrier of iteration
        // i+1 orders that read before the overwrite.
    }

    // ---- epilogue: normalize, write [B*S][1024] f16 (col = h*64 + d) ----
    int b = bh >> 4;
    int h = bh & 15;
#pragma unroll
    for (int r = 0; r < 8; ++r) {
        float inv  = 1.0f / rl[r];
        int   srow = q0 + r + mh;
        size_t base = ((size_t)(b * SS + srow)) * DM + h * DK + nn;
        ao[base +  0] = (_Float16)(o0[r] * inv);
        ao[base + 16] = (_Float16)(o1[r] * inv);
        ao[base + 32] = (_Float16)(o2[r] * inv);
        ao[base + 48] = (_Float16)(o3[r] * inv);
    }
}

// ---------------------------------------------------------------------------
// Host-side launch
// ---------------------------------------------------------------------------
extern "C" void kernel_launch(void* const* d_in, const int* in_sizes, int n_in,
                              void* d_out, int out_size, void* d_ws, size_t ws_size,
                              hipStream_t stream) {
    const float* x     = (const float*)d_in[0];   // [B,S,1024] f32
    const int*   tpos  = (const int*)d_in[1];     // [B,S] i32
    const float* qkv_w = (const float*)d_in[2];   // [3072,1024] f32
    const float* out_w = (const float*)d_in[3];   // [1024,1024] f32
    float*       out   = (float*)d_out;           // [B,S,1024] f32

    char* ws = (char*)d_ws;
    _Float16* xh  = (_Float16*)(ws);                         //  8 MB
    _Float16* wqh = (_Float16*)(ws + 8388608);               //  6 MB
    _Float16* woh = (_Float16*)(ws + 14680064);              //  2 MB
    float*    qkv = (float*)   (ws + 16777216);              // 48 MB
    _Float16* qh  = (_Float16*)(ws + 67108864);              //  8 MB
    _Float16* kh  = (_Float16*)(ws + 75497472);              //  8 MB
    _Float16* vT  = (_Float16*)(ws + 83886080);              //  8 MB
    _Float16* ao  = (_Float16*)(ws + 92274688);              //  8 MB (total 96 MB)

    // 1) convert inputs to f16
    f32_to_f16_kernel<<<2048, 256, 0, stream>>>(x,     xh,  ROWS * DM);
    f32_to_f16_kernel<<<2048, 256, 0, stream>>>(qkv_w, wqh, E3 * DM);
    f32_to_f16_kernel<<<1024, 256, 0, stream>>>(out_w, woh, DM * DM);

    // 2) QKV projection: 64x48 = 3072 waves -> 768 blocks of 4 waves
    gemm_nt_kernel<<<768, 128, 0, stream>>>(xh, wqh, qkv, ROWS, E3, DM);

    // 3) RoPE + split into q/k (rotated, head-major) and v (transposed)
    rope_split_kernel<<<(BB * SS * HH * (DK / 2)) / 256, 256, 0, stream>>>(
        qkv, tpos, qh, kh, vT);

    // 4) flash attention: 32 heads * 32 blocks (64 q-rows each)
    attn_kernel<<<1024, 128, 0, stream>>>(qh, kh, vT, ao);

    // 5) output projection: 64x16 = 1024 waves -> 256 blocks
    gemm_nt_kernel<<<256, 128, 0, stream>>>(ao, woh, out, ROWS, DM, DM);
}
